// RelativeMultiHeadAttention_10642928959687
// MI455X (gfx1250) — matrix operand order repaired and checked by reading, not compiled, not run on hardware
//
#include <hip/hip_runtime.h>
#include <hip/hip_bf16.h>

// ---------------------------------------------------------------------------
// RelativeMultiHeadAttention forward for MI455X (gfx1250, wave32, WMMA).
// B=2, S=1024, HS=1024, H=16, D=64, vocab=1025.
// All GEMMs use v_wmma_f32_16x16x32_bf16 (bf16 operands, fp32 accumulate).
// Softmax / LayerNorm / attention-weights output stay fp32.
// Logits are written directly into the d_out "weights" slot and softmaxed
// in place to avoid a 128 MB workspace round trip (memory-bound problem).
// ---------------------------------------------------------------------------

#define BATCH 2
#define SEQ   1024
#define HSZ   1024
#define NH    16
#define HD    64

typedef __attribute__((ext_vector_type(16))) __bf16 v16bf;
typedef __attribute__((ext_vector_type(8)))  __bf16 v8bf;
typedef __attribute__((ext_vector_type(8)))  float  v8f;

__device__ __forceinline__ v8f wmma_bf16(v16bf a, v16bf b, v8f c) {
  return __builtin_amdgcn_wmma_f32_16x16x32_bf16(
      /*neg_a=*/false, a, /*neg_b=*/false, b,
      /*c_mod=*/(short)0, c, /*reuse_a=*/false, /*reuse_b=*/false);
}

__device__ __forceinline__ v8bf ld8(const __bf16* p) {
  return *(const v8bf*)p;                 // 16B load
}

__device__ __forceinline__ v8bf cvt8(const float* p) {
  v8f f = *(const v8f*)p;                 // 32B load
  v8bf r;
#pragma unroll
  for (int i = 0; i < 8; ++i) r[i] = (__bf16)f[i];
  return r;
}

__device__ __forceinline__ v16bf cat(v8bf lo, v8bf hi) {
  return __builtin_shufflevector(lo, hi, 0, 1, 2, 3, 4, 5, 6, 7,
                                 8, 9, 10, 11, 12, 13, 14, 15);
}

// A fragment (16 rows x 32 K), contiguous K order used by this file (B matches):
// lane&15 = row; lanes<16: K {0..15}; lanes>=16: K {16..31}.
__device__ __forceinline__ v16bf a_frag_bf16(const __bf16* base, int ld,
                                             int m0, int k0, int lane) {
  int hf = lane >> 4, r = lane & 15;
  const __bf16* p = base + (size_t)(m0 + r) * ld + k0 + hf * 16;
  return cat(ld8(p), ld8(p + 8));
}

__device__ __forceinline__ v16bf a_frag_f32(const float* base, int ld,
                                            int m0, int k0, int lane) {
  int hf = lane >> 4, r = lane & 15;
  const float* p = base + (size_t)(m0 + r) * ld + k0 + hf * 16;
  return cat(cvt8(p), cvt8(p + 8));
}

// B fragment: lane holds column lane&15; 16 contiguous K values starting at p
// (p already includes k0 + (lane>>4)*16 for the column's contiguous run).
__device__ __forceinline__ v16bf b_frag_bf16(const __bf16* p) {
  return cat(ld8(p), ld8(p + 8));
}
__device__ __forceinline__ v16bf b_frag_f32(const float* p) {
  return cat(cvt8(p), cvt8(p + 8));
}

// ---------------------------------------------------------------------------
// K1: fused Q/K/V projections. out = X @ W^T + b, split heads.
//   qh,kh : bf16 (B,H,S,D)   vhT : bf16 (B,H,D,S)  (transposed for ctx GEMM)
// One wave computes a 32x32 tile; which in {0,1,2} selects Q/K/V.
// ---------------------------------------------------------------------------
__global__ __launch_bounds__(256)
void qkv_proj_kernel(const float* __restrict__ xq, const float* __restrict__ xk,
                     const float* __restrict__ xv,
                     const float* __restrict__ Wq, const float* __restrict__ bq,
                     const float* __restrict__ Wk, const float* __restrict__ bk,
                     const float* __restrict__ Wv, const float* __restrict__ bv,
                     __bf16* __restrict__ qh, __bf16* __restrict__ kh,
                     __bf16* __restrict__ vhT) {
  int gw   = (blockIdx.x * blockDim.x + threadIdx.x) >> 5;
  int lane = threadIdx.x & 31;
  const int MT = (BATCH * SEQ) / 32;  // 64
  const int NT = HSZ / 32;            // 32
  int which = gw / (MT * NT);
  int rem   = gw % (MT * NT);
  int m0 = (rem / NT) * 32;
  int n0 = (rem % NT) * 32;

  const float* X    = (which == 0) ? xq : (which == 1) ? xk : xv;
  const float* W    = (which == 0) ? Wq : (which == 1) ? Wk : Wv;
  const float* bias = (which == 0) ? bq : (which == 1) ? bk : bv;

  int hf = lane >> 4, col = lane & 15;
  v8f c[2][2] = {};
  for (int k0 = 0; k0 < HSZ; k0 += 32) {
    __builtin_prefetch(X + (size_t)(m0 + col) * HSZ + k0 + 64, 0, 0);
    v16bf a[2], b[2];
#pragma unroll
    for (int i = 0; i < 2; ++i)
      a[i] = a_frag_f32(X, HSZ, m0 + i * 16, k0, lane);
#pragma unroll
    for (int j = 0; j < 2; ++j)
      b[j] = b_frag_f32(W + (size_t)(n0 + j * 16 + col) * HSZ + k0 + hf * 16);
#pragma unroll
    for (int i = 0; i < 2; ++i)
#pragma unroll
      for (int j = 0; j < 2; ++j) c[i][j] = wmma_bf16(a[i], b[j], c[i][j]);
  }

#pragma unroll
  for (int j = 0; j < 2; ++j) {
    int n  = n0 + j * 16 + col;      // output feature = h*HD + d
    float bb = bias[n];
    int h = n >> 6, d = n & 63;
#pragma unroll
    for (int i = 0; i < 2; ++i) {
#pragma unroll
      for (int r = 0; r < 8; ++r) {
        int m = m0 + i * 16 + r + hf * 8;   // flat (b,s)
        int b_ = m >> 10, s = m & 1023;
        __bf16 val = (__bf16)(c[i][j][r] + bb);
        if (which == 2)
          vhT[(((size_t)(b_ * NH + h)) * HD + d) * SEQ + s] = val;
        else if (which == 0)
          qh[(((size_t)(b_ * NH + h)) * SEQ + s) * HD + d] = val;
        else
          kh[(((size_t)(b_ * NH + h)) * SEQ + s) * HD + d] = val;
      }
    }
  }
}

// ---------------------------------------------------------------------------
// K2: content logits: for each (b,h): L[q,k] = qh[q,:] . kh[k,:]  (K=64)
// One wave computes a 32x32 tile of L, written fp32 into the weights slot.
// ---------------------------------------------------------------------------
__global__ __launch_bounds__(256)
void content_logits_kernel(const __bf16* __restrict__ qh,
                           const __bf16* __restrict__ kh,
                           float* __restrict__ logits) {
  int gw   = (blockIdx.x * blockDim.x + threadIdx.x) >> 5;
  int lane = threadIdx.x & 31;
  int bh  = gw / (32 * 32);
  int rem = gw % (32 * 32);
  int q0  = (rem / 32) * 32;
  int k0c = (rem % 32) * 32;

  const __bf16* Qb = qh + (size_t)bh * SEQ * HD;
  const __bf16* Kb = kh + (size_t)bh * SEQ * HD;
  int hf = lane >> 4, col = lane & 15;

  v8f c[2][2] = {};
#pragma unroll
  for (int d0 = 0; d0 < HD; d0 += 32) {
    v16bf a[2], b[2];
#pragma unroll
    for (int i = 0; i < 2; ++i) a[i] = a_frag_bf16(Qb, HD, q0 + i * 16, d0, lane);
#pragma unroll
    for (int j = 0; j < 2; ++j)
      b[j] = b_frag_bf16(Kb + (size_t)(k0c + j * 16 + col) * HD + d0 + hf * 16);
#pragma unroll
    for (int i = 0; i < 2; ++i)
#pragma unroll
      for (int j = 0; j < 2; ++j) c[i][j] = wmma_bf16(a[i], b[j], c[i][j]);
  }

  float* Lb = logits + (size_t)bh * SEQ * SEQ;
#pragma unroll
  for (int i = 0; i < 2; ++i)
#pragma unroll
    for (int r = 0; r < 8; ++r) {
      size_t row = (size_t)(q0 + i * 16 + r + hf * 8) * SEQ;
#pragma unroll
      for (int j = 0; j < 2; ++j)
        Lb[row + k0c + j * 16 + col] = c[i][j][r];
    }
}

// ---------------------------------------------------------------------------
// K3: relative-key logits. For fixed q: (32 bh x 64) @ (64 x S) where
// B column k = rel_key_emb[rel_positions[q,k], :]. Adds into logits.
// One wave: fixed q, one 16-wide k tile, both 16-row bh tiles (M=32).
// ---------------------------------------------------------------------------
__global__ __launch_bounds__(256)
void rel_logits_kernel(const __bf16* __restrict__ qh,
                       const int* __restrict__ relpos,
                       const float* __restrict__ rke,
                       float* __restrict__ logits) {
  int gw   = (blockIdx.x * blockDim.x + threadIdx.x) >> 5;
  int lane = threadIdx.x & 31;
  int qi = gw / (SEQ / 16);
  int k0 = (gw % (SEQ / 16)) * 16;
  int hf = lane >> 4, col = lane & 15;

  int rp = relpos[(size_t)qi * SEQ + k0 + col];
  const float* erow = rke + (size_t)rp * HD;

  v8f c[2] = {};
#pragma unroll
  for (int d0 = 0; d0 < HD; d0 += 32) {
    v16bf b = b_frag_f32(erow + d0 + hf * 16);
    v16bf a0 = a_frag_bf16(qh + (size_t)qi * HD, SEQ * HD, 0,  d0, lane);
    v16bf a1 = a_frag_bf16(qh + (size_t)qi * HD, SEQ * HD, 16, d0, lane);
    c[0] = wmma_bf16(a0, b, c[0]);
    c[1] = wmma_bf16(a1, b, c[1]);
  }
#pragma unroll
  for (int i = 0; i < 2; ++i)
#pragma unroll
    for (int r = 0; r < 8; ++r) {
      int bh = i * 16 + r + hf * 8;
      size_t idx = ((size_t)bh * SEQ + qi) * SEQ + k0 + col;
      logits[idx] += c[i][r];
    }
}

// ---------------------------------------------------------------------------
// K4: row softmax over k, fp32 in place (d_out weights slot). 1 block / row.
// ---------------------------------------------------------------------------
__global__ __launch_bounds__(256)
void softmax_kernel(float* __restrict__ logits) {
  __shared__ float smax[8], ssum[8];
  float* p = logits + (size_t)blockIdx.x * SEQ;
  int t = threadIdx.x;
  float4 x = ((const float4*)p)[t];
  float m = fmaxf(fmaxf(x.x, x.y), fmaxf(x.z, x.w));
#pragma unroll
  for (int o = 16; o > 0; o >>= 1) m = fmaxf(m, __shfl_xor(m, o, 32));
  if ((t & 31) == 0) smax[t >> 5] = m;
  __syncthreads();
  float bm = smax[0];
#pragma unroll
  for (int i = 1; i < 8; ++i) bm = fmaxf(bm, smax[i]);
  float e0 = expf(x.x - bm), e1 = expf(x.y - bm);
  float e2 = expf(x.z - bm), e3 = expf(x.w - bm);
  float s = e0 + e1 + e2 + e3;
#pragma unroll
  for (int o = 16; o > 0; o >>= 1) s += __shfl_xor(s, o, 32);
  if ((t & 31) == 0) ssum[t >> 5] = s;
  __syncthreads();
  float tot = 0.f;
#pragma unroll
  for (int i = 0; i < 8; ++i) tot += ssum[i];
  float inv = 1.0f / tot;
  ((float4*)p)[t] = make_float4(e0 * inv, e1 * inv, e2 * inv, e3 * inv);
}

// ---------------------------------------------------------------------------
// K5: content context: for each (b,h): ctx[q,d] = sum_k w[q,k]*vh[k,d].
// A = softmaxed weights (fp32 -> bf16 on load), B columns from vhT (B,H,D,S).
// One wave: 32 q x 64 d, K = S. Writes ctx fp32 ('=').
// ---------------------------------------------------------------------------
__global__ __launch_bounds__(256)
void ctx_content_kernel(const float* __restrict__ wgt,
                        const __bf16* __restrict__ vhT,
                        float* __restrict__ ctx) {
  int gw   = (blockIdx.x * blockDim.x + threadIdx.x) >> 5;
  int lane = threadIdx.x & 31;
  int bh = gw / 32;
  int q0 = (gw % 32) * 32;
  int hf = lane >> 4, col = lane & 15;

  const float* Wb  = wgt + (size_t)bh * SEQ * SEQ;
  const __bf16* Vb = vhT + (size_t)bh * HD * SEQ;

  v8f c[2][4] = {};
  for (int k0 = 0; k0 < SEQ; k0 += 32) {
    __builtin_prefetch(Wb + (size_t)(q0 + col) * SEQ + k0 + 64, 0, 0);
    v16bf a[2], b[4];
#pragma unroll
    for (int i = 0; i < 2; ++i) a[i] = a_frag_f32(Wb, SEQ, q0 + i * 16, k0, lane);
#pragma unroll
    for (int nt = 0; nt < 4; ++nt)
      b[nt] = b_frag_bf16(Vb + (size_t)(nt * 16 + col) * SEQ + k0 + hf * 16);
#pragma unroll
    for (int i = 0; i < 2; ++i)
#pragma unroll
      for (int nt = 0; nt < 4; ++nt) c[i][nt] = wmma_bf16(a[i], b[nt], c[i][nt]);
  }

#pragma unroll
  for (int i = 0; i < 2; ++i)
#pragma unroll
    for (int r = 0; r < 8; ++r) {
      size_t row = ((size_t)bh * SEQ + q0 + i * 16 + r + hf * 8) * HD;
#pragma unroll
      for (int nt = 0; nt < 4; ++nt)
        ctx[row + nt * 16 + col] = c[i][nt][r];
    }
}

// ---------------------------------------------------------------------------
// K6: relative-value context. For fixed q: (32 bh x S) @ (S x 64) where
// B[k][d] = rel_val_emb[rel_positions[q,k], d]. Adds into ctx.
// One wave per q; rel_positions row staged in LDS (per-wave slice, no barrier).
// ---------------------------------------------------------------------------
__global__ __launch_bounds__(256)
void ctx_rel_kernel(const float* __restrict__ wgt,
                    const int* __restrict__ relpos,
                    const float* __restrict__ rve,
                    float* __restrict__ ctx) {
  __shared__ int rp_s[8][SEQ];
  int wv   = threadIdx.x >> 5;
  int lane = threadIdx.x & 31;
  int qi = blockIdx.x * 8 + wv;
  int hf = lane >> 4, col = lane & 15;

  // stage this wave's rel_positions row into LDS
  const int4* src = (const int4*)(relpos + (size_t)qi * SEQ);
  int4* dst = (int4*)rp_s[wv];
  for (int i = lane; i < SEQ / 4; i += 32) dst[i] = src[i];

  v8f c[2][4] = {};
  for (int k0 = 0; k0 < SEQ; k0 += 32) {
    v16bf a[2];
#pragma unroll
    for (int i = 0; i < 2; ++i)   // rows = bh, stride S*S between bh rows
      a[i] = a_frag_f32(wgt + (size_t)qi * SEQ, SEQ * SEQ, i * 16, k0, lane);
    int kkb = k0 + hf * 16;
    v16bf b[4];
#pragma unroll
    for (int nt = 0; nt < 4; ++nt) {
      int d = nt * 16 + col;
      v16bf bb;
#pragma unroll
      for (int e = 0; e < 16; ++e) {
        int idx = rp_s[wv][kkb + e];
        bb[e] = (__bf16)rve[(size_t)idx * HD + d];
      }
      b[nt] = bb;
    }
#pragma unroll
    for (int i = 0; i < 2; ++i)
#pragma unroll
      for (int nt = 0; nt < 4; ++nt) c[i][nt] = wmma_bf16(a[i], b[nt], c[i][nt]);
  }

#pragma unroll
  for (int i = 0; i < 2; ++i)
#pragma unroll
    for (int r = 0; r < 8; ++r) {
      int bh = i * 16 + r + hf * 8;
      size_t row = ((size_t)bh * SEQ + qi) * HD;
#pragma unroll
      for (int nt = 0; nt < 4; ++nt)
        ctx[row + nt * 16 + col] += c[i][nt][r];
    }
}

// ---------------------------------------------------------------------------
// K7: output projection: y = ctx_flat @ Wo^T + bo, ctx_flat[m, h*HD+d] taken
// from ctx (B,H,S,D). One wave: 32x32 tile. y fp32 to workspace.
// ---------------------------------------------------------------------------
__global__ __launch_bounds__(256)
void out_proj_kernel(const float* __restrict__ ctx,
                     const float* __restrict__ Wo,
                     const float* __restrict__ bo,
                     float* __restrict__ y) {
  int gw   = (blockIdx.x * blockDim.x + threadIdx.x) >> 5;
  int lane = threadIdx.x & 31;
  const int NT = HSZ / 32;  // 32
  int m0 = (gw / NT) * 32;
  int n0 = (gw % NT) * 32;
  int hf = lane >> 4, col = lane & 15;

  v8f c[2][2] = {};
  for (int k0 = 0; k0 < HSZ; k0 += 32) {
    v16bf a[2], b[2];
#pragma unroll
    for (int i = 0; i < 2; ++i) {
      int m = m0 + i * 16 + col;
      int b_ = m >> 10, s = m & 1023;
      int c0 = k0 + hf * 16, c1 = c0 + 8;
      const float* p0 =
          ctx + (((size_t)(b_ * NH) + (c0 >> 6)) * SEQ + s) * HD + (c0 & 63);
      const float* p1 =
          ctx + (((size_t)(b_ * NH) + (c1 >> 6)) * SEQ + s) * HD + (c1 & 63);
      a[i] = cat(cvt8(p0), cvt8(p1));
    }
#pragma unroll
    for (int j = 0; j < 2; ++j)
      b[j] = b_frag_f32(Wo + (size_t)(n0 + j * 16 + col) * HSZ + k0 + hf * 16);
#pragma unroll
    for (int i = 0; i < 2; ++i)
#pragma unroll
      for (int j = 0; j < 2; ++j) c[i][j] = wmma_bf16(a[i], b[j], c[i][j]);
  }

#pragma unroll
  for (int j = 0; j < 2; ++j) {
    int n = n0 + j * 16 + col;
    float bb = bo[n];
#pragma unroll
    for (int i = 0; i < 2; ++i)
#pragma unroll
      for (int r = 0; r < 8; ++r) {
        int m = m0 + i * 16 + r + hf * 8;
        y[(size_t)m * HSZ + n] = c[i][j][r] + bb;
      }
  }
}

// ---------------------------------------------------------------------------
// K8: residual add + LayerNorm over last dim (1024). 1 block / row.
// ---------------------------------------------------------------------------
__global__ __launch_bounds__(256)
void resid_ln_kernel(const float* __restrict__ y, const float* __restrict__ resid,
                     const float* __restrict__ g, const float* __restrict__ b,
                     float* __restrict__ out) {
  __shared__ float s1[8], s2[8];
  size_t row = blockIdx.x;
  int t = threadIdx.x;
  float4 yy = ((const float4*)(y + row * HSZ))[t];
  float4 rr = ((const float4*)(resid + row * HSZ))[t];
  float4 xv = make_float4(yy.x + rr.x, yy.y + rr.y, yy.z + rr.z, yy.w + rr.w);

  float ls = xv.x + xv.y + xv.z + xv.w;
#pragma unroll
  for (int o = 16; o > 0; o >>= 1) ls += __shfl_xor(ls, o, 32);
  if ((t & 31) == 0) s1[t >> 5] = ls;
  __syncthreads();
  float tot = 0.f;
#pragma unroll
  for (int i = 0; i < 8; ++i) tot += s1[i];
  float mu = tot * (1.0f / HSZ);

  float d0 = xv.x - mu, d1 = xv.y - mu, d2 = xv.z - mu, d3 = xv.w - mu;
  float lv = d0 * d0 + d1 * d1 + d2 * d2 + d3 * d3;
#pragma unroll
  for (int o = 16; o > 0; o >>= 1) lv += __shfl_xor(lv, o, 32);
  if ((t & 31) == 0) s2[t >> 5] = lv;
  __syncthreads();
  float tv = 0.f;
#pragma unroll
  for (int i = 0; i < 8; ++i) tv += s2[i];
  float inv = rsqrtf(tv * (1.0f / HSZ) + 1e-5f);

  float4 gg = ((const float4*)g)[t];
  float4 bb = ((const float4*)b)[t];
  ((float4*)(out + row * HSZ))[t] =
      make_float4(d0 * inv * gg.x + bb.x, d1 * inv * gg.y + bb.y,
                  d2 * inv * gg.z + bb.z, d3 * inv * gg.w + bb.w);
}

// ---------------------------------------------------------------------------
extern "C" void kernel_launch(void* const* d_in, const int* in_sizes, int n_in,
                              void* d_out, int out_size, void* d_ws,
                              size_t ws_size, hipStream_t stream) {
  (void)in_sizes; (void)n_in; (void)out_size; (void)ws_size;
  const float* q   = (const float*)d_in[0];
  const float* k   = (const float*)d_in[1];
  const float* v   = (const float*)d_in[2];
  const int* relp  = (const int*)d_in[3];
  const float* Wq  = (const float*)d_in[4];
  const float* bq  = (const float*)d_in[5];
  const float* Wk  = (const float*)d_in[6];
  const float* bk  = (const float*)d_in[7];
  const float* Wv  = (const float*)d_in[8];
  const float* bv  = (const float*)d_in[9];
  const float* rke = (const float*)d_in[10];
  const float* rve = (const float*)d_in[11];
  const float* Wo  = (const float*)d_in[12];
  const float* bo  = (const float*)d_in[13];
  const float* lng = (const float*)d_in[14];
  const float* lnb = (const float*)d_in[15];

  float* out     = (float*)d_out;                       // (B,S,HS) = 2M f32
  float* weights = out + (size_t)BATCH * SEQ * HSZ;     // (B,H,S,S) = 32M f32

  // workspace: 28 MB total
  char* ws = (char*)d_ws;
  __bf16* qh  = (__bf16*)(ws);                          //  4 MB (B,H,S,D)
  __bf16* kh  = (__bf16*)(ws + (4u << 20));             //  4 MB (B,H,S,D)
  __bf16* vhT = (__bf16*)(ws + (8u << 20));             //  4 MB (B,H,D,S)
  float*  ctx = (float*) (ws + (12u << 20));            //  8 MB (B,H,S,D)
  float*  yb  = (float*) (ws + (20u << 20));            //  8 MB (B,S,HS)

  // K1: 3 * 64 * 32 tiles = 6144 waves -> 768 blocks
  qkv_proj_kernel<<<768, 256, 0, stream>>>(q, k, v, Wq, bq, Wk, bk, Wv, bv,
                                           qh, kh, vhT);
  // K2: 32 * 32 * 32 = 32768 waves -> 4096 blocks
  content_logits_kernel<<<4096, 256, 0, stream>>>(qh, kh, weights);
  // K3: 1024 * 64 = 65536 waves -> 8192 blocks
  rel_logits_kernel<<<8192, 256, 0, stream>>>(qh, relp, rke, weights);
  // K4: 32768 rows
  softmax_kernel<<<32768, 256, 0, stream>>>(weights);
  // K5: 32 * 32 = 1024 waves -> 128 blocks
  ctx_content_kernel<<<128, 256, 0, stream>>>(weights, vhT, ctx);
  // K6: 1024 q, 8 waves/block -> 128 blocks
  ctx_rel_kernel<<<128, 256, 0, stream>>>(weights, relp, rve, ctx);
  // K7: 64 * 32 = 2048 waves -> 256 blocks
  out_proj_kernel<<<256, 256, 0, stream>>>(ctx, Wo, bo, yb);
  // K8: 2048 rows
  resid_ln_kernel<<<2048, 256, 0, stream>>>(yb, q, lng, lnb, out);
}
